// MultiHeadLinearAttention_61220463837902
// MI455X (gfx1250) — matrix operand order repaired and checked by reading, not compiled, not run on hardware
//
#include <hip/hip_runtime.h>
#include <hip/hip_bf16.h>

typedef __attribute__((ext_vector_type(16))) _Float16 v16h;
typedef __attribute__((ext_vector_type(8)))  _Float16 v8h;
typedef __attribute__((ext_vector_type(4)))  _Float16 v4h;
typedef __attribute__((ext_vector_type(8)))  float    v8f;
typedef __attribute__((ext_vector_type(4)))  float    v4f;

#define NB    4
#define NH    16
#define DH    64
#define LSEQ  4096
#define EDIM  1024
#define CHUNK 256     // K rows per block in pass 1
#define KSTEP 32      // WMMA K depth (f16)
#define NIT   (CHUNK / KSTEP)
#define P1    (KSTEP + 8)   // 40 halves = 80B row pitch (16B-aligned rows)
#define P2    (DH + 8)      // 72 halves = 144B row pitch (16B-aligned rows)

// sigmoid(0.6053*x - 4.102) = rcp(1 + exp2((4.102 - 0.6053*x) * log2(e)))
#define PHI_A (-0.873263308258529f)   // -0.6053 * log2(e)
#define PHI_B ( 5.917935057726374f)   //  4.102  * log2(e)

__device__ __forceinline__ float phi_sig(float x) {
    const float t = __builtin_fmaf(x, PHI_A, PHI_B);
    return __builtin_amdgcn_rcpf(1.0f + __builtin_amdgcn_exp2f(t));
}

__device__ __forceinline__ v8f wmma_f16(const v16h& a, const v16h& b, const v8f& c) {
    return __builtin_amdgcn_wmma_f32_16x16x32_f16(false, a, false, b, (short)0, c,
                                                  false, false);
}

// Build 16x32 A fragment from k-contiguous LDS row (two 16B ds_load_b128)
__device__ __forceinline__ v16h frag_a(const _Float16* rowPtr, int hlf) {
    const v8h lo = *(const v8h*)(rowPtr + hlf * 8);        // K = hlf*8 .. +7
    const v8h hi = *(const v8h*)(rowPtr + 16 + hlf * 8);   // K = 16+hlf*8 .. +7
    v16h a;
#pragma unroll
    for (int j = 0; j < 8; ++j) { a[j] = lo[j]; a[8 + j] = hi[j]; }
    return a;
}

// Build 32x16 B fragment from k-contiguous LDS row (two 16B ds_load_b128)
__device__ __forceinline__ v16h frag_b(const _Float16* rowPtr, int hlf) {
    const v8h lo = *(const v8h*)(rowPtr + hlf * 8);        // K = hlf*8 .. +7
    const v8h hi = *(const v8h*)(rowPtr + 16 + hlf * 8);   // K = 16+hlf*8 .. +7
    v16h b;
#pragma unroll
    for (int j = 0; j < 8; ++j) { b[j] = lo[j]; b[8 + j] = hi[j]; }
    return b;
}

// ---------------------------------------------------------------------------
// Pass 1: kv[bh][d][e] = sum_k phi(K[b,k,h*64+d]) * V[b,k,h*64+e]
//         ksum[bh][d]  = sum_k phi(K[b,k,h*64+d])
// grid: (NB*NH, LSEQ/CHUNK), block: 256 threads (8 waves)
// ---------------------------------------------------------------------------
__global__ void __launch_bounds__(256)
mla_kv_kernel(const float* __restrict__ K, const float* __restrict__ V,
              float* __restrict__ kv, float* __restrict__ ksum) {
    const int bh   = blockIdx.x;
    const int b    = bh / NH;
    const int h    = bh % NH;
    const int row0 = blockIdx.y * CHUNK;

    const int tid    = threadIdx.x;
    const int wave   = tid >> 5;
    const int lane   = tid & 31;
    const int hlf    = lane >> 4;   // 0 or 1
    const int lane16 = lane & 15;

    // Double-buffered, transposed (k-contiguous) staging buffers
    __shared__ __align__(16) _Float16 sKd[2][DH][P1];   // phi(K)^T : [d][k]
    __shared__ __align__(16) _Float16 sVt[2][DH][P1];   // V^T      : [e][k]

    const int dt  = wave & 3;          // d-tile (A shared by both accumulators)
    const int et0 = (wave >> 2) * 2;   // e-tiles et0, et0+1

    v8f acc0 = {};
    v8f acc1 = {};

    const float* Kbase = K + ((size_t)b * LSEQ) * EDIM + h * DH;
    const float* Vbase = V + ((size_t)b * LSEQ) * EDIM + h * DH;

    // 4x4 register-transpose staging: waves 0-3 handle K(+phi), waves 4-7 V
    const bool doK  = (tid < 128);
    const int  blk  = tid & 127;        // 128 4x4 blocks per matrix
    const int  bc   = blk & 15;         // col block -> cols bc*4..+3
    const int  br   = blk >> 4;         // row block -> rows br*4..+3 (of 32)
    const float* Gbase = doK ? Kbase : Vbase;

    // ksum partials in registers (K-side waves only), 4 independent chains
    float ksp[4] = {0.f, 0.f, 0.f, 0.f};

    // Software pipeline: prefetch chunk 0 (non-temporal: single-use stream)
    v4f g[4];
#pragma unroll
    for (int i = 0; i < 4; ++i)
        g[i] = __builtin_nontemporal_load(
            (const v4f*)(Gbase + (size_t)(row0 + br * 4 + i) * EDIM + bc * 4));

    for (int it = 0; it < NIT; ++it) {
        const int pb = it & 1;
        // One wave-uniform branch per iteration (NOT per element): keeps phi
        // off the V path without per-element saveexec ladders.
        if (doK) {
#pragma unroll
            for (int c = 0; c < 4; ++c) {
                v4h p;
#pragma unroll
                for (int i = 0; i < 4; ++i) {
                    const float pf = phi_sig(g[i][c]);
                    ksp[c] += pf;
                    p[i] = (_Float16)pf;
                }
                *(v4h*)&sKd[pb][bc * 4 + c][br * 4] = p;   // ds_store_b64
            }
        } else {
#pragma unroll
            for (int c = 0; c < 4; ++c) {
                v4h p;
#pragma unroll
                for (int i = 0; i < 4; ++i) p[i] = (_Float16)g[i][c];
                *(v4h*)&sVt[pb][bc * 4 + c][br * 4] = p;   // ds_store_b64
            }
        }
        __syncthreads();   // single barrier/iter: double buffer makes it safe

        // Issue next chunk's HBM loads now: overlap with WMMA + next staging
        if (it + 1 < NIT) {
#pragma unroll
            for (int i = 0; i < 4; ++i)
                g[i] = __builtin_nontemporal_load(
                    (const v4f*)(Gbase +
                        (size_t)(row0 + (it + 1) * KSTEP + br * 4 + i) * EDIM + bc * 4));
        }

        const v16h a  = frag_a(&sKd[pb][dt * 16 + lane16][0], hlf);
        const v16h b0 = frag_b(&sVt[pb][(et0 + 0) * 16 + lane16][0], hlf);
        const v16h b1 = frag_b(&sVt[pb][(et0 + 1) * 16 + lane16][0], hlf);
        acc0 = wmma_f16(a, b0, acc0);
        acc1 = wmma_f16(a, b1, acc1);
    }

    // Accumulate partial kv into workspace (coalesced f32 atomics), [d][e]
    float* kvb = kv + (size_t)bh * DH * DH;
#pragma unroll
    for (int i = 0; i < 8; ++i) {
        const int m = dt * 16 + i + 8 * hlf;
        atomicAdd(&kvb[m * DH + (et0 + 0) * 16 + lane16], acc0[i]);
        atomicAdd(&kvb[m * DH + (et0 + 1) * 16 + lane16], acc1[i]);
    }
    if (doK) {
#pragma unroll
        for (int c = 0; c < 4; ++c)
            atomicAdd(&ksum[bh * DH + bc * 4 + c], ksp[c]);
    }
}

// ---------------------------------------------------------------------------
// Pass 2: out[b,q,h*64+e] = (phi(Q) @ kv) / (phi(Q) @ ksum)
// grid: (NB*NH, LSEQ/64), block: 256 threads (8 waves)
// ---------------------------------------------------------------------------
__global__ void __launch_bounds__(256)
mla_out_kernel(const float* __restrict__ Q, const float* __restrict__ kv,
               const float* __restrict__ ksum, float* __restrict__ out) {
    const int bh = blockIdx.x;
    const int b  = bh / NH;
    const int h  = bh % NH;
    const int q0 = blockIdx.y * 64;

    const int tid    = threadIdx.x;
    const int wave   = tid >> 5;
    const int lane   = tid & 31;
    const int hlf    = lane >> 4;
    const int lane16 = lane & 15;

    __shared__ __align__(16) _Float16 sQ[64][P2];     // phi(Q) : [q][d] (d-contig)
    __shared__ __align__(16) _Float16 sKVt[DH][P2];   // kv^T   : [e][d] (d-contig)
    __shared__ float sSum[DH];
    __shared__ float sDen[64];

    const float* kvb   = kv + (size_t)bh * DH * DH;               // [d][e]
    const float* Qbase = Q + ((size_t)b * LSEQ + q0) * EDIM + h * DH;
#pragma unroll
    for (int i = 0; i < 4; ++i) {
        const int idx = tid + i * 256;     // 0..1023
        const int r   = idx >> 4;          // row 0..63
        const int c4  = idx & 15;
        // phi(Q): streaming (non-temporal) load; 4 contiguous halves store
        const v4f qv = __builtin_nontemporal_load(
            (const v4f*)(Qbase + (size_t)r * EDIM + c4 * 4));
        v4h qp;
#pragma unroll
        for (int c = 0; c < 4; ++c) qp[c] = (_Float16)phi_sig(qv[c]);
        *(v4h*)&sQ[r][c4 * 4] = qp;
        // kv: regular-temporal (L2-resident, reused by 64 blocks); transpose
        const v4f kvv = *(const v4f*)(kvb + r * DH + c4 * 4);
        sKVt[c4 * 4 + 0][r] = (_Float16)kvv[0];
        sKVt[c4 * 4 + 1][r] = (_Float16)kvv[1];
        sKVt[c4 * 4 + 2][r] = (_Float16)kvv[2];
        sKVt[c4 * 4 + 3][r] = (_Float16)kvv[3];
    }
    if (tid < DH) sSum[tid] = ksum[bh * DH + tid];
    __syncthreads();

    // Denominator: den[q] = sum_d phiQ[q][d]*ksum[d]
    // (waves 0..1; 4 independent partials to break the serial FMA chain)
    if (tid < 64) {
        float p0 = 0.0f, p1 = 0.0f, p2 = 0.0f, p3 = 0.0f;
        const v8h q0v = *(const v8h*)&sQ[tid][0];
        const v8h q1v = *(const v8h*)&sQ[tid][8];
        const v8h q2v = *(const v8h*)&sQ[tid][16];
        const v8h q3v = *(const v8h*)&sQ[tid][24];
        const v8h q4v = *(const v8h*)&sQ[tid][32];
        const v8h q5v = *(const v8h*)&sQ[tid][40];
        const v8h q6v = *(const v8h*)&sQ[tid][48];
        const v8h q7v = *(const v8h*)&sQ[tid][56];
#pragma unroll
        for (int i = 0; i < 8; ++i) {
            p0 += (float)q0v[i] * sSum[0 + i] ;
            p1 += (float)q1v[i] * sSum[8 + i] ;
            p2 += (float)q2v[i] * sSum[16 + i];
            p3 += (float)q3v[i] * sSum[24 + i];
            p0 += (float)q4v[i] * sSum[32 + i];
            p1 += (float)q5v[i] * sSum[40 + i];
            p2 += (float)q6v[i] * sSum[48 + i];
            p3 += (float)q7v[i] * sSum[56 + i];
        }
        sDen[tid] = (p0 + p1) + (p2 + p3);
    }

    // Numerator via WMMA: 16 tiles over 8 waves, K = 64 in 2 steps of 32
    const int qt  = wave & 3;
    const int et0 = (wave >> 2) * 2;
    v8f acc0 = {};
    v8f acc1 = {};
#pragma unroll
    for (int it = 0; it < 2; ++it) {
        const v16h a  = frag_a(&sQ[qt * 16 + lane16][it * 32], hlf);
        const v16h b0 = frag_b(&sKVt[(et0 + 0) * 16 + lane16][it * 32], hlf);
        const v16h b1 = frag_b(&sKVt[(et0 + 1) * 16 + lane16][it * 32], hlf);
        acc0 = wmma_f16(a, b0, acc0);
        acc1 = wmma_f16(a, b1, acc1);
    }
    __syncthreads();   // sDen visible

    float* ob = out + ((size_t)b * LSEQ + q0) * EDIM + h * DH;
#pragma unroll
    for (int i = 0; i < 8; ++i) {
        const int m   = qt * 16 + i + 8 * hlf;
        const float r = __builtin_amdgcn_rcpf(sDen[m]);   // v_rcp_f32
        __builtin_nontemporal_store(acc0[i] * r,
            ob + (size_t)m * EDIM + (et0 + 0) * 16 + lane16);
        __builtin_nontemporal_store(acc1[i] * r,
            ob + (size_t)m * EDIM + (et0 + 1) * 16 + lane16);
    }
}

extern "C" void kernel_launch(void* const* d_in, const int* in_sizes, int n_in,
                              void* d_out, int out_size, void* d_ws, size_t ws_size,
                              hipStream_t stream) {
    const float* Q = (const float*)d_in[0];
    const float* K = (const float*)d_in[1];
    const float* V = (const float*)d_in[2];
    float* out = (float*)d_out;

    float* kv   = (float*)d_ws;                       // [64][64][64] f32
    float* ksum = kv + (size_t)NB * NH * DH * DH;     // [64][64]     f32
    const size_t ws_used = ((size_t)NB * NH * DH * DH + (size_t)NB * NH * DH) * sizeof(float);

    // Atomically-accumulated state must start at zero every call (capture-safe)
    hipMemsetAsync(d_ws, 0, ws_used, stream);

    mla_kv_kernel<<<dim3(NB * NH, LSEQ / CHUNK), 256, 0, stream>>>(K, V, kv, ksum);
    mla_out_kernel<<<dim3(NB * NH, LSEQ / 64), 256, 0, stream>>>(Q, kv, ksum, out);
}
